// GraphFormerBlock_1864015806552
// MI455X (gfx1250) — matrix operand order repaired and checked by reading, not compiled, not run on hardware
//
#include <hip/hip_runtime.h>
#include <hip/hip_bf16.h>

// ---------------------------------------------------------------------------
// GraphFormer block on MI455X (gfx1250, wave32).
// GEMMs: V_WMMA_F32_16X16X32_F16, 4 wmma per wave per K-step (16x64 strip).
// All LDS staging and fragment traffic is b128 (weights pre-transposed to
// [N,K] f16 so the B tile stages without scalar transpose stores).
// A-side operands (x, h) pre-converted to f16 to halve streamed HBM bytes.
// Edge softmax via 3-pass atomics; beta-gate + BN + ELU fused per node.
// ---------------------------------------------------------------------------

typedef __attribute__((ext_vector_type(16))) _Float16 v16h;
typedef __attribute__((ext_vector_type(8)))  _Float16 v8h;
typedef __attribute__((ext_vector_type(8)))  float    v8f;
typedef __attribute__((ext_vector_type(4)))  float    v4f;

#define IN_DIM 128
#define D1     256
#define HEADS  4
#define HID    64
#define OUTD   256
#define N0_    262144
#define N1_    65536
#define N2_    4096
#define E1_    655360
#define E2_    65536
#define BN_EPS 1e-5f

// ------------------------- helpers -----------------------------------------

__device__ __forceinline__ float waveReduceSum(float v) {
  #pragma unroll
  for (int o = 16; o > 0; o >>= 1) v += __shfl_xor(v, o, 32);
  return v;
}

// order-preserving float <-> uint encoding for atomicMax on signed floats
__device__ __forceinline__ unsigned fenc(float f) {
  unsigned u = __float_as_uint(f);
  return (u >> 31) ? ~u : (u | 0x80000000u);
}
__device__ __forceinline__ float fdec(unsigned u) {
  return __uint_as_float((u >> 31) ? (u & 0x7FFFFFFFu) : ~u);
}
#define ENC_NEG_INF 0x007FFFFFu   // fenc(-inf)

__global__ void fill_u32_kernel(unsigned* __restrict__ p, unsigned v, int n) {
  int i = blockIdx.x * 256 + threadIdx.x;
  if (i < n) p[i] = v;
}

// 4-wide f32 -> f16 convert (n % 4 == 0 for all users)
__global__ __launch_bounds__(256)
void f32_to_f16_kernel(const float* __restrict__ in, _Float16* __restrict__ out,
                       int n) {
  int i = (blockIdx.x * 256 + threadIdx.x) * 4;
  if (i >= n) return;
  const v4f x = *(const v4f*)(in + i);
  out[i + 0] = (_Float16)x.x;
  out[i + 1] = (_Float16)x.y;
  out[i + 2] = (_Float16)x.z;
  out[i + 3] = (_Float16)x.w;
}

// f32 [K,N] -> f16 transposed [N,K]; weights only (small, converted once)
__global__ __launch_bounds__(256)
void f32_to_f16_T_kernel(const float* __restrict__ in, _Float16* __restrict__ out,
                         int K, int N) {
  int idx = blockIdx.x * 256 + threadIdx.x;
  if (idx >= K * N) return;
  const int k = idx / N, n = idx - k * N;   // coalesced read along N
  out[(size_t)n * K + k] = (_Float16)in[idx];
}

// ------------------------- WMMA GEMM ----------------------------------------
// C[M,N] = A[M,K] @ W[K,N] + bias[N]; A f16 [M,K], Wt f16 [N,K] (transposed).
// Block: 256 thr = 8 waves. Block tile 128(M) x 64(N); wave strip 16x64.
// A LDS layout: [m][k] natural order (no swizzle); each lane-half's
//   16 values (K 0-15 / K 16-31) are contiguous (2x ds_load_b128).
// B LDS layout: [n][k] (natural for Wt); lane-half -> contiguous K 0-15/16-31.
// Requirements: M%128==0, N%64==0, K%32==0 (all shapes satisfy).

#define MT 128
#define NT 64
#define KT 32
#define ASTR 40   // f16 units per A row (80B, mult of 16B, conflict-light)
#define BSTR 40   // f16 units per B row

__global__ __launch_bounds__(256)
void gemm_wmma_f16(const _Float16* __restrict__ A, const _Float16* __restrict__ Wt,
                   const float* __restrict__ bias, float* __restrict__ C,
                   int M, int K, int N) {
  __shared__ __align__(16) _Float16 As[MT][ASTR];
  __shared__ __align__(16) _Float16 Bt[NT][BSTR];

  const int tid     = threadIdx.x;
  const int wave    = tid >> 5;           // 0..7
  const int lane    = tid & 31;
  const int rowBase = blockIdx.y * MT;
  const int colBase = blockIdx.x * NT;

  v8f acc0 = {}, acc1 = {}, acc2 = {}, acc3 = {};

  for (int k0 = 0; k0 < K; k0 += KT) {
    // stage A tile 128x32: thread t -> row t/2, K-half (t&1)*16 (16 f16 = 32B)
    {
      const int r    = tid >> 1;
      const int half = tid & 1;
      const _Float16* src = A + (size_t)(rowBase + r) * K + k0 + half * 16;
      const v8h g0 = *(const v8h*)(src);
      const v8h g1 = *(const v8h*)(src + 8);
      // natural order: half 0: c0-7 -> pos0, c8-15 -> pos8
      //                half 1: c16-23 -> pos16, c24-31 -> pos24
      *(v8h*)&As[r][half * 16]     = g0;
      *(v8h*)&As[r][half * 16 + 8] = g1;
    }
    // stage B tile (64 n) x (32 k) from Wt: thread t -> n = t/4, kp = (t&3)*8
    {
      const int n  = tid >> 2;
      const int kp = (tid & 3) * 8;
      const v8h w = *(const v8h*)(Wt + (size_t)(colBase + n) * K + k0 + kp);
      *(v8h*)&Bt[n][kp] = w;
    }
    __syncthreads();

    // fragments: two ds_load_b128 each; preload all B frags, then WMMA chain
    const int r    = wave * 16 + (lane & 15);
    const int asel = (lane >> 4) * 16;    // lane-half -> contiguous group
    const int ko   = (lane >> 4) * 16;
    const int nl   = lane & 15;
    const v16h a = __builtin_shufflevector(
        *(const v8h*)&As[r][asel], *(const v8h*)&As[r][asel + 8],
        0,1,2,3,4,5,6,7,8,9,10,11,12,13,14,15);
    const v16h b0 = __builtin_shufflevector(
        *(const v8h*)&Bt[0*16 + nl][ko], *(const v8h*)&Bt[0*16 + nl][ko + 8],
        0,1,2,3,4,5,6,7,8,9,10,11,12,13,14,15);
    const v16h b1 = __builtin_shufflevector(
        *(const v8h*)&Bt[1*16 + nl][ko], *(const v8h*)&Bt[1*16 + nl][ko + 8],
        0,1,2,3,4,5,6,7,8,9,10,11,12,13,14,15);
    const v16h b2 = __builtin_shufflevector(
        *(const v8h*)&Bt[2*16 + nl][ko], *(const v8h*)&Bt[2*16 + nl][ko + 8],
        0,1,2,3,4,5,6,7,8,9,10,11,12,13,14,15);
    const v16h b3 = __builtin_shufflevector(
        *(const v8h*)&Bt[3*16 + nl][ko], *(const v8h*)&Bt[3*16 + nl][ko + 8],
        0,1,2,3,4,5,6,7,8,9,10,11,12,13,14,15);
    acc0 = __builtin_amdgcn_wmma_f32_16x16x32_f16(false, a, false, b0,
                                                  (short)0, acc0, false, false);
    acc1 = __builtin_amdgcn_wmma_f32_16x16x32_f16(false, a, false, b1,
                                                  (short)0, acc1, false, false);
    acc2 = __builtin_amdgcn_wmma_f32_16x16x32_f16(false, a, false, b2,
                                                  (short)0, acc2, false, false);
    acc3 = __builtin_amdgcn_wmma_f32_16x16x32_f16(false, a, false, b3,
                                                  (short)0, acc3, false, false);
    __syncthreads();
  }

  // store: acc[r] -> (M = r + 8*(lane/16), N = lane%16) per 16x16 subtile
  const int nl = lane & 15;
  const int mh = (lane >> 4) * 8;
  #pragma unroll
  for (int j = 0; j < 4; ++j) {
    const v8f acc = (j == 0) ? acc0 : (j == 1) ? acc1 : (j == 2) ? acc2 : acc3;
    const int col = colBase + j * 16 + nl;
    const float bv = bias[col];
    #pragma unroll
    for (int r = 0; r < 8; ++r) {
      const int row = rowBase + wave * 16 + mh + r;
      C[(size_t)row * N + col] = acc[r] + bv;
    }
  }
}

// -------------------- layer 1 edge attention (H=4, C=64) -------------------

__global__ __launch_bounds__(256)
void alpha1_kernel(const float* __restrict__ q, const float* __restrict__ k,
                   const int* __restrict__ src, const int* __restrict__ dst,
                   float* __restrict__ ealpha, unsigned* __restrict__ amax,
                   int EH) {
  const int eh = blockIdx.x * 8 + (threadIdx.x >> 5);
  if (eh >= EH) return;
  const int lane = threadIdx.x & 31;
  const int e = eh >> 2, h = eh & 3;
  const int s = src[e], d = dst[e];
  const float* qp = q + (size_t)d * D1 + h * HID;
  const float* kp = k + (size_t)s * D1 + h * HID;
  float acc = qp[lane * 2] * kp[lane * 2] + qp[lane * 2 + 1] * kp[lane * 2 + 1];
  acc = waveReduceSum(acc);
  if (lane == 0) {
    const float a = acc * 0.125f;                 // 1/sqrt(64)
    ealpha[eh] = a;
    atomicMax(amax + (size_t)d * HEADS + h, fenc(a));
  }
}

__global__ __launch_bounds__(256)
void expsum1_kernel(float* __restrict__ ealpha, const unsigned* __restrict__ amax,
                    float* __restrict__ denom, const int* __restrict__ dst,
                    int EH) {
  const int tid = blockIdx.x * 256 + threadIdx.x;
  if (tid >= EH) return;
  const int e = tid >> 2, h = tid & 3;
  const int d = dst[e];
  const float ex = __expf(ealpha[tid] - fdec(amax[(size_t)d * HEADS + h]));
  ealpha[tid] = ex;
  atomicAdd(denom + (size_t)d * HEADS + h, ex);
}

__global__ __launch_bounds__(256)
void scatter1_kernel(const float* __restrict__ ealpha, const float* __restrict__ denom,
                     const float* __restrict__ v, const int* __restrict__ src,
                     const int* __restrict__ dst, float* __restrict__ agg) {
  const int e = blockIdx.x, t = threadIdx.x;
  const int s = src[e], d = dst[e];
  const int h = t >> 6;
  const float attn = ealpha[(size_t)e * HEADS + h] / denom[(size_t)d * HEADS + h];
  atomicAdd(agg + (size_t)d * D1 + t, attn * v[(size_t)s * D1 + t]);
}

// -------------------- layer 2 edge attention (H=1, C=256) ------------------

__global__ __launch_bounds__(256)
void alpha2_kernel(const float* __restrict__ q, const float* __restrict__ k,
                   const int* __restrict__ src, const int* __restrict__ dst,
                   float* __restrict__ ealpha, unsigned* __restrict__ amax, int E) {
  const int e = blockIdx.x * 8 + (threadIdx.x >> 5);
  if (e >= E) return;
  const int lane = threadIdx.x & 31;
  const int s = src[e], d = dst[e];
  const float* qp = q + (size_t)d * OUTD;
  const float* kp = k + (size_t)s * OUTD;
  float acc = 0.0f;
  #pragma unroll
  for (int j = 0; j < 8; ++j) acc += qp[lane + 32 * j] * kp[lane + 32 * j];
  acc = waveReduceSum(acc);
  if (lane == 0) {
    const float a = acc * 0.0625f;                // 1/sqrt(256)
    ealpha[e] = a;
    atomicMax(amax + d, fenc(a));
  }
}

__global__ __launch_bounds__(256)
void expsum2_kernel(float* __restrict__ ealpha, const unsigned* __restrict__ amax,
                    float* __restrict__ denom, const int* __restrict__ dst, int E) {
  const int e = blockIdx.x * 256 + threadIdx.x;
  if (e >= E) return;
  const int d = dst[e];
  const float ex = __expf(ealpha[e] - fdec(amax[d]));
  ealpha[e] = ex;
  atomicAdd(denom + d, ex);
}

__global__ __launch_bounds__(256)
void scatter2_kernel(const float* __restrict__ ealpha, const float* __restrict__ denom,
                     const float* __restrict__ v, const int* __restrict__ src,
                     const int* __restrict__ dst, float* __restrict__ agg) {
  const int e = blockIdx.x, t = threadIdx.x;
  const int s = src[e], d = dst[e];
  const float attn = ealpha[e] / denom[d];
  atomicAdd(agg + (size_t)d * OUTD + t, attn * v[(size_t)s * OUTD + t]);
}

// -------------------- beta gate (+ optional BN + ELU) ----------------------
// One block (256 threads) per node; channel t = threadIdx.x.
// Writes f32 (outf) and/or f16 (outh) -- layer-1 writes f16 h for the
// layer-2 WMMA GEMMs; the final layer writes f32 d_out.

__global__ __launch_bounds__(256)
void beta_gate_kernel(const float* __restrict__ agg, const float* __restrict__ xr,
                      const float* __restrict__ Wbeta,   // [3*256] column vector
                      const float* __restrict__ gamma, const float* __restrict__ bbeta,
                      const float* __restrict__ mean, const float* __restrict__ var,
                      float* __restrict__ outf, _Float16* __restrict__ outh,
                      int doBnElu) {
  __shared__ float red[8];
  const int i = blockIdx.x, t = threadIdx.x;
  const float o = agg[(size_t)i * 256 + t];
  const float r = xr[(size_t)i * 256 + t];
  float p = o * Wbeta[t] + r * Wbeta[256 + t] + (o - r) * Wbeta[512 + t];
  p = waveReduceSum(p);
  if ((t & 31) == 0) red[t >> 5] = p;
  __syncthreads();
  const float s = red[0] + red[1] + red[2] + red[3] +
                  red[4] + red[5] + red[6] + red[7];
  const float beta = 1.0f / (1.0f + __expf(-s));
  float hv = beta * r + (1.0f - beta) * o;
  if (doBnElu) {
    hv = (hv - mean[t]) * gamma[t] * rsqrtf(var[t] + BN_EPS) + bbeta[t];
    hv = (hv > 0.0f) ? hv : expm1f(hv);           // ELU(alpha=1)
  }
  if (outf) outf[(size_t)i * 256 + t] = hv;
  if (outh) outh[(size_t)i * 256 + t] = (_Float16)hv;
}

// --------------------------- launcher --------------------------------------

extern "C" void kernel_launch(void* const* d_in, const int* in_sizes, int n_in,
                              void* d_out, int out_size, void* d_ws, size_t ws_size,
                              hipStream_t stream) {
  (void)in_sizes; (void)n_in; (void)out_size; (void)ws_size;

  const float* x      = (const float*)d_in[0];
  const int*   esrc1  = (const int*)d_in[1];
  const int*   edst1  = (const int*)d_in[2];
  const int*   esrc2  = (const int*)d_in[3];
  const int*   edst2  = (const int*)d_in[4];
  const float* Wq1 = (const float*)d_in[5];  const float* bq1 = (const float*)d_in[6];
  const float* Wk1 = (const float*)d_in[7];  const float* bk1 = (const float*)d_in[8];
  const float* Wv1 = (const float*)d_in[9];  const float* bv1 = (const float*)d_in[10];
  const float* Ws1 = (const float*)d_in[11]; const float* bs1 = (const float*)d_in[12];
  const float* Wbeta1 = (const float*)d_in[13];
  const float* Wq2 = (const float*)d_in[14]; const float* bq2 = (const float*)d_in[15];
  const float* Wk2 = (const float*)d_in[16]; const float* bk2 = (const float*)d_in[17];
  const float* Wv2 = (const float*)d_in[18]; const float* bv2 = (const float*)d_in[19];
  const float* Ws2 = (const float*)d_in[20]; const float* bs2 = (const float*)d_in[21];
  const float* Wbeta2   = (const float*)d_in[22];
  const float* bn_gamma = (const float*)d_in[23];
  const float* bn_beta  = (const float*)d_in[24];
  const float* bn_mean  = (const float*)d_in[25];
  const float* bn_var   = (const float*)d_in[26];

  // ---- workspace layout (float units; all segments 16B aligned) ----
  float* ws = (float*)d_ws;
  size_t off = 0;
  float* k1   = ws + off; off += (size_t)N0_ * D1;     // 67.1M
  float* v1   = ws + off; off += (size_t)N0_ * D1;     // 67.1M
  float* q1   = ws + off; off += (size_t)N1_ * D1;
  float* s1   = ws + off; off += (size_t)N1_ * D1;
  float* agg1 = ws + off; off += (size_t)N1_ * D1;
  float* ea1  = ws + off; off += (size_t)E1_ * HEADS;
  unsigned* amax1 = (unsigned*)(ws + off); off += (size_t)N1_ * HEADS;
  float* den1 = ws + off; off += (size_t)N1_ * HEADS;
  // f16 pool (sizes in f16 units, offsets advanced in float units = n/2)
  _Float16* xh  = (_Float16*)(ws + off); off += (size_t)N0_ * IN_DIM / 2;
  _Float16* hh  = (_Float16*)(ws + off); off += (size_t)N1_ * D1 / 2;
  _Float16* w1h = (_Float16*)(ws + off); off += (size_t)4 * IN_DIM * D1 / 2;
  _Float16* w2h = (_Float16*)(ws + off); off += (size_t)4 * D1 * OUTD / 2;
  _Float16* Wq1t = w1h + 0 * IN_DIM * D1, *Wk1t = w1h + 1 * IN_DIM * D1;
  _Float16* Wv1t = w1h + 2 * IN_DIM * D1, *Ws1t = w1h + 3 * IN_DIM * D1;
  _Float16* Wq2t = w2h + 0 * D1 * OUTD,   *Wk2t = w2h + 1 * D1 * OUTD;
  _Float16* Wv2t = w2h + 2 * D1 * OUTD,   *Ws2t = w2h + 3 * D1 * OUTD;
  // layer-2 buffers alias the k1 region (k1 dead after alpha1 pass)
  float* k2   = k1;
  float* v2   = k2 + (size_t)N1_ * OUTD;
  float* q2   = v2 + (size_t)N1_ * OUTD;
  float* s2   = q2 + (size_t)N2_ * OUTD;
  float* agg2 = s2 + (size_t)N2_ * OUTD;
  float* ea2  = agg2 + (size_t)N2_ * OUTD;
  unsigned* amax2 = (unsigned*)(ea2 + E2_);
  float* den2 = (float*)(amax2 + N2_);

  const dim3 blk(256);
  #define CVT(src, dst, n) \
    hipLaunchKernelGGL(f32_to_f16_kernel, dim3(((n) / 4 + 255) / 256), blk, 0, \
                       stream, src, dst, n)
  #define CVTT(src, dst, Kd, Nd) \
    hipLaunchKernelGGL(f32_to_f16_T_kernel, dim3(((Kd) * (Nd) + 255) / 256), blk, \
                       0, stream, src, dst, Kd, Nd)

  // ---- f16 conversions (x plain; 8 weight matrices transposed to [N,K]) ----
  CVT(x, xh, N0_ * IN_DIM);
  CVTT(Wq1, Wq1t, IN_DIM, D1);  CVTT(Wk1, Wk1t, IN_DIM, D1);
  CVTT(Wv1, Wv1t, IN_DIM, D1);  CVTT(Ws1, Ws1t, IN_DIM, D1);
  CVTT(Wq2, Wq2t, D1, OUTD);    CVTT(Wk2, Wk2t, D1, OUTD);
  CVTT(Wv2, Wv2t, D1, OUTD);    CVTT(Ws2, Ws2t, D1, OUTD);

  // ---- layer 1 projections ----
  { dim3 g(D1 / NT, N1_ / MT);
    hipLaunchKernelGGL(gemm_wmma_f16, g, blk, 0, stream, xh, Wq1t, bq1, q1, N1_, IN_DIM, D1); }
  { dim3 g(D1 / NT, N0_ / MT);
    hipLaunchKernelGGL(gemm_wmma_f16, g, blk, 0, stream, xh, Wk1t, bk1, k1, N0_, IN_DIM, D1); }
  { dim3 g(D1 / NT, N0_ / MT);
    hipLaunchKernelGGL(gemm_wmma_f16, g, blk, 0, stream, xh, Wv1t, bv1, v1, N0_, IN_DIM, D1); }
  { dim3 g(D1 / NT, N1_ / MT);
    hipLaunchKernelGGL(gemm_wmma_f16, g, blk, 0, stream, xh, Ws1t, bs1, s1, N1_, IN_DIM, D1); }

  // ---- layer 1 segment softmax + scatter ----
  { const int n = N1_ * HEADS;
    hipLaunchKernelGGL(fill_u32_kernel, dim3((n + 255) / 256), blk, 0, stream,
                       amax1, ENC_NEG_INF, n); }
  hipMemsetAsync(den1, 0, (size_t)N1_ * HEADS * sizeof(float), stream);
  hipMemsetAsync(agg1, 0, (size_t)N1_ * D1 * sizeof(float), stream);

  { const int EH = E1_ * HEADS;
    hipLaunchKernelGGL(alpha1_kernel, dim3((EH + 7) / 8), blk, 0, stream,
                       q1, k1, esrc1, edst1, ea1, amax1, EH);
    hipLaunchKernelGGL(expsum1_kernel, dim3((EH + 255) / 256), blk, 0, stream,
                       ea1, amax1, den1, edst1, EH); }
  hipLaunchKernelGGL(scatter1_kernel, dim3(E1_), blk, 0, stream,
                     ea1, den1, v1, esrc1, edst1, agg1);

  // ---- beta gate + BN + ELU -> h (f16 for layer-2 GEMMs) ----
  hipLaunchKernelGGL(beta_gate_kernel, dim3(N1_), blk, 0, stream,
                     agg1, s1, Wbeta1, bn_gamma, bn_beta, bn_mean, bn_var,
                     (float*)nullptr, hh, 1);

  // ---- layer 2 projections ----
  { dim3 g(OUTD / NT, N2_ / MT);
    hipLaunchKernelGGL(gemm_wmma_f16, g, blk, 0, stream, hh, Wq2t, bq2, q2, N2_, D1, OUTD); }
  { dim3 g(OUTD / NT, N1_ / MT);
    hipLaunchKernelGGL(gemm_wmma_f16, g, blk, 0, stream, hh, Wk2t, bk2, k2, N1_, D1, OUTD); }
  { dim3 g(OUTD / NT, N1_ / MT);
    hipLaunchKernelGGL(gemm_wmma_f16, g, blk, 0, stream, hh, Wv2t, bv2, v2, N1_, D1, OUTD); }
  { dim3 g(OUTD / NT, N2_ / MT);
    hipLaunchKernelGGL(gemm_wmma_f16, g, blk, 0, stream, hh, Ws2t, bs2, s2, N2_, D1, OUTD); }

  // ---- layer 2 segment softmax + scatter ----
  hipLaunchKernelGGL(fill_u32_kernel, dim3((N2_ + 255) / 256), blk, 0, stream,
                     amax2, ENC_NEG_INF, N2_);
  hipMemsetAsync(den2, 0, (size_t)N2_ * sizeof(float), stream);
  hipMemsetAsync(agg2, 0, (size_t)N2_ * OUTD * sizeof(float), stream);

  hipLaunchKernelGGL(alpha2_kernel, dim3((E2_ + 7) / 8), blk, 0, stream,
                     q2, k2, esrc2, edst2, ea2, amax2, E2_);
  hipLaunchKernelGGL(expsum2_kernel, dim3((E2_ + 255) / 256), blk, 0, stream,
                     ea2, amax2, den2, edst2, E2_);
  hipLaunchKernelGGL(scatter2_kernel, dim3(E2_), blk, 0, stream,
                     ea2, den2, v2, esrc2, edst2, agg2);

  // ---- final beta gate -> d_out (f32) ----
  hipLaunchKernelGGL(beta_gate_kernel, dim3(N2_), blk, 0, stream,
                     agg2, s2, Wbeta2, (const float*)nullptr, (const float*)nullptr,
                     (const float*)nullptr, (const float*)nullptr,
                     (float*)d_out, (_Float16*)nullptr, 0);
  #undef CVT
  #undef CVTT
}
